// GraphConvDist_8650064134752
// MI455X (gfx1250) — compile-verified, matrix operand order checked by reading, not run on hardware
//
#include <hip/hip_runtime.h>
#include <hip/hip_bf16.h>

typedef __attribute__((ext_vector_type(16))) __bf16 v16bf;
typedef __attribute__((ext_vector_type(8)))  __bf16 v8bf;
typedef __attribute__((ext_vector_type(8)))  float  v8f;

#define H_DIM    128
#define F_IN     384
#define N_NODES  65536
#define E_EDGES  1000000
#define EDGE_TILE 64
#define NWAVES    4

// workspace layout
#define AGG_BYTES ((size_t)N_NODES * H_DIM * 4)            // 32 MB fp32 agg
#define W1P_OFF   AGG_BYTES
#define W1P_BYTES ((size_t)12 * 8 * 32 * 16 * 2)           // 96 KB bf16 packed W1
#define W2P_OFF   (W1P_OFF + W1P_BYTES)                    // 32 KB bf16 packed W2

#define NEG_INIT_BITS 0xFF7FFFFFu                          // bits of -FLT_MAX

// ---------------------------------------------------------------- init agg
__global__ void init_agg_kernel(uint4* __restrict__ agg) {
    size_t i = (size_t)blockIdx.x * blockDim.x + threadIdx.x;
    uint4 v = { NEG_INIT_BITS, NEG_INIT_BITS, NEG_INIT_BITS, NEG_INIT_BITS };
    agg[i] = v;
}

// ------------------------------------------------- pack weights to B-fragments
// Fragment order (16x32 bf16 B tile, wave32):
//   lane l: g = l>>4, r = l&15 ; column N = 16*n + r
//   element e (0..15): K = 32*c + 8*g + (e < 8 ? e : e + 8)
// One contiguous 32B run per lane -> v16bf load in the GEMM.
__global__ void pack_w_kernel(const float* __restrict__ W, __bf16* __restrict__ Wp,
                              int nchunks) {
    int tid = blockIdx.x * blockDim.x + threadIdx.x;
    int total = nchunks * 8 * 32;
    if (tid >= total) return;
    int l = tid & 31;
    int n = (tid >> 5) & 7;
    int c = tid >> 8;
    int g = l >> 4, r = l & 15;
    int col = 16 * n + r;
    __bf16* dst = Wp + (size_t)tid * 16;
#pragma unroll
    for (int e = 0; e < 16; ++e) {
        int k = 32 * c + 8 * g + (e < 8 ? e : e + 8);
        dst[e] = (__bf16)W[k * H_DIM + col];
    }
}

// ---------------------------------------------------- edge MLP + scatter max
__global__ __launch_bounds__(128) void edge_mlp_scatter_kernel(
    const float* __restrict__ X,        // [E, 384] fp32
    const int*   __restrict__ node_idx, // [E]
    const float* __restrict__ b1,       // [128]
    const float* __restrict__ b2,       // [128]
    const __bf16* __restrict__ W1p,     // packed fragments, 12 chunks
    const __bf16* __restrict__ W2p,     // packed fragments, 4 chunks
    float* __restrict__ agg)            // [N, 128] init'd to -FLT_MAX bits
{
    __shared__ __bf16 H1[NWAVES][16][H_DIM];  // per-wave relu'd hidden strip

    const int w    = threadIdx.x >> 5;
    const int lane = threadIdx.x & 31;
    const int g    = lane >> 4;
    const int r    = lane & 15;

    const size_t edgeRow = (size_t)blockIdx.x * EDGE_TILE + w * 16 + r;
    const float* xrow = X + edgeRow * F_IN;

    const v16bf* W1f = (const v16bf*)W1p;
    const v16bf* W2f = (const v16bf*)W2p;

    // ---------------- layer 1: [16x384] @ [384x128] in 12 K-chunks of 32
    v8f acc[8] = {};
#pragma unroll
    for (int c = 0; c < 12; ++c) {
        const int kb = 32 * c + 8 * g;
        float4 x0 = *(const float4*)(xrow + kb);
        float4 x1 = *(const float4*)(xrow + kb + 4);
        float4 x2 = *(const float4*)(xrow + kb + 16);
        float4 x3 = *(const float4*)(xrow + kb + 20);
        v16bf a;
        a[0]=(__bf16)x0.x; a[1]=(__bf16)x0.y; a[2]=(__bf16)x0.z; a[3]=(__bf16)x0.w;
        a[4]=(__bf16)x1.x; a[5]=(__bf16)x1.y; a[6]=(__bf16)x1.z; a[7]=(__bf16)x1.w;
        a[8]=(__bf16)x2.x; a[9]=(__bf16)x2.y; a[10]=(__bf16)x2.z; a[11]=(__bf16)x2.w;
        a[12]=(__bf16)x3.x; a[13]=(__bf16)x3.y; a[14]=(__bf16)x3.z; a[15]=(__bf16)x3.w;
#pragma unroll
        for (int n = 0; n < 8; ++n) {
            v16bf b = W1f[(c * 8 + n) * 32 + lane];
            acc[n] = __builtin_amdgcn_wmma_f32_16x16x32_bf16(
                false, a, false, b, (short)0, acc[n], false, false);
        }
    }

    // bias + relu -> bf16 strip in LDS (own wave's region only)
#pragma unroll
    for (int n = 0; n < 8; ++n) {
        float bb = b1[16 * n + r];
#pragma unroll
        for (int v = 0; v < 8; ++v) {
            float val = fmaxf(acc[n][v] + bb, 0.0f);
            H1[w][8 * g + v][16 * n + r] = (__bf16)val;
        }
    }

    // ---------------- layer 2: [16x128] @ [128x128] in 4 K-chunks of 32
    v8f acc2[8] = {};
#pragma unroll
    for (int c = 0; c < 4; ++c) {
        const int kb = 32 * c + 8 * g;
        union { v16bf v; v8bf h[2]; } a2;
        a2.h[0] = *(const v8bf*)&H1[w][r][kb];
        a2.h[1] = *(const v8bf*)&H1[w][r][kb + 16];
#pragma unroll
        for (int n = 0; n < 8; ++n) {
            v16bf b = W2f[(c * 8 + n) * 32 + lane];
            acc2[n] = __builtin_amdgcn_wmma_f32_16x16x32_bf16(
                false, a2.v, false, b, (short)0, acc2[n], false, false);
        }
    }

    // ---------------- bias + scatter-max (hardware int/uint atomics, no CAS)
    const int ebase = blockIdx.x * EDGE_TILE + w * 16 + 8 * g;
    int idxs[8];
#pragma unroll
    for (int v = 0; v < 8; ++v) idxs[v] = node_idx[ebase + v];

#pragma unroll
    for (int n = 0; n < 8; ++n) {
        const int col = 16 * n + r;
        const float bb = b2[col];
#pragma unroll
        for (int v = 0; v < 8; ++v) {
            float val = acc2[n][v] + bb;
            float* addr = agg + (size_t)idxs[v] * H_DIM + col;
            if (val >= 0.0f)
                atomicMax((int*)addr, __float_as_int(val));
            else
                atomicMin((unsigned int*)addr, __float_as_uint(val));
        }
    }
}

// ----------------------------------------------------------- cosine finalize
__global__ void finalize_kernel(const float* __restrict__ center,
                                const float* __restrict__ gcn,
                                const float* __restrict__ agg,
                                float* __restrict__ out) {
    int row  = blockIdx.x * 8 + (threadIdx.x >> 5);
    int lane = threadIdx.x & 31;
    size_t base = (size_t)row * H_DIM + lane * 4;

    float4 gv = *(const float4*)(gcn + base);
    float4 av = *(const float4*)(agg + base);
    float4 cv = *(const float4*)(center + base);

    if (__float_as_uint(av.x) == NEG_INIT_BITS) av.x = 0.0f;
    if (__float_as_uint(av.y) == NEG_INIT_BITS) av.y = 0.0f;
    if (__float_as_uint(av.z) == NEG_INIT_BITS) av.z = 0.0f;
    if (__float_as_uint(av.w) == NEG_INIT_BITS) av.w = 0.0f;

    float4 lv = { cv.x + av.x, cv.y + av.y, cv.z + av.z, cv.w + av.w };

    float dot = gv.x * lv.x + gv.y * lv.y + gv.z * lv.z + gv.w * lv.w;
    float na  = gv.x * gv.x + gv.y * gv.y + gv.z * gv.z + gv.w * gv.w;
    float nb  = lv.x * lv.x + lv.y * lv.y + lv.z * lv.z + lv.w * lv.w;

#pragma unroll
    for (int off = 16; off > 0; off >>= 1) {
        dot += __shfl_xor(dot, off, 32);
        na  += __shfl_xor(na,  off, 32);
        nb  += __shfl_xor(nb,  off, 32);
    }
    if (lane == 0) {
        float denom = fmaxf(sqrtf(na), 1e-8f) * fmaxf(sqrtf(nb), 1e-8f);
        out[row] = dot / denom;
    }
}

// ---------------------------------------------------------------------------
extern "C" void kernel_launch(void* const* d_in, const int* in_sizes, int n_in,
                              void* d_out, int out_size, void* d_ws, size_t ws_size,
                              hipStream_t stream) {
    const float* center = (const float*)d_in[0];
    const float* leaf   = (const float*)d_in[1];
    const int*   nidx   = (const int*)d_in[2];
    const float* gcn    = (const float*)d_in[3];
    const float* W1     = (const float*)d_in[4];
    const float* b1     = (const float*)d_in[5];
    const float* W2     = (const float*)d_in[6];
    const float* b2     = (const float*)d_in[7];
    float* out = (float*)d_out;

    char* ws = (char*)d_ws;
    float*  agg = (float*)ws;
    __bf16* W1p = (__bf16*)(ws + W1P_OFF);
    __bf16* W2p = (__bf16*)(ws + W2P_OFF);

    // init agg to -FLT_MAX bit pattern (N*H/4 uint4 = 8192 * 256 threads)
    init_agg_kernel<<<(N_NODES * H_DIM / 4) / 256, 256, 0, stream>>>((uint4*)agg);

    // pack weights into per-lane WMMA fragment order (bf16)
    pack_w_kernel<<<12, 256, 0, stream>>>(W1, W1p, 12);
    pack_w_kernel<<<4,  256, 0, stream>>>(W2, W2p, 4);

    // edge MLP + scatter-max: E = 15625 * 64 exactly
    edge_mlp_scatter_kernel<<<E_EDGES / EDGE_TILE, 128, 0, stream>>>(
        leaf, nidx, b1, b2, W1p, W2p, agg);

    // cosine similarity per node row
    finalize_kernel<<<N_NODES / 8, 256, 0, stream>>>(center, gcn, agg, out);
}
